// ModernBertAttention_28372553957583
// MI455X (gfx1250) — matrix operand order repaired and checked by reading, not compiled, not run on hardware
//
#include <hip/hip_runtime.h>
#include <stdint.h>

// ModernBERT sliding-window attention for gfx1250 (MI455X), wave32 + WMMA.
// Pipeline: rope/pack(f16) -> weight transpose(f16) -> QKV wmma GEMMs
// (32x64 register-blocked, weights double-buffer-staged to LDS via
// global_load_async_to_lds_b128 / s_wait_asynccnt) -> flash sliding-window
// attention (wmma QK^T + online softmax + wmma P*V) -> output wmma GEMM (f32).

#define S_LEN 2048
#define NHEAD 12
#define HDIM  64
#define HID   768
#define BATCH 4
#define WIN   128

typedef __attribute__((ext_vector_type(16))) _Float16 v16h;
typedef __attribute__((ext_vector_type(8)))  float    v8f;

union FragU { v16h v; uint32_t u[8]; };

// A-fragment (16x32 f16, M x K), element (row,k) at tile[row*ld + k].
// ISA 7.12.2: lane l holds row M=l&15; lanes 0-15 K=0..7,16..23; lanes 16-31 K=8..15,24..31.
__device__ __forceinline__ v16h load_frag_a(const _Float16* tile, int ld, int kbase) {
  int lane = threadIdx.x & 31;
  int row  = lane & 15;
  int h    = lane >> 4;
  FragU f;
  const _Float16* p = tile + (size_t)row * ld + kbase;
#pragma unroll
  for (int g = 0; g < 2; ++g)
#pragma unroll
    for (int r2 = 0; r2 < 4; ++r2)
      f.u[g * 4 + r2] = *(const uint32_t*)(p + g * 16 + h * 8 + r2 * 2);
  return f.v;
}

// B-fragment (32x16 f16, K x N) from K-contiguous (transposed) storage:
// element (k,n) at baseT[(nbase+n)*ld + k]. N=lane&15; lanes 0-15 K=0..7,16..23; lanes 16-31 K=8..15,24..31.
__device__ __forceinline__ v16h load_frag_b_t(const _Float16* baseT, int ld, int kbase, int nbase) {
  int lane = threadIdx.x & 31;
  int n    = lane & 15;
  int h    = lane >> 4;
  FragU f;
  const _Float16* p = baseT + (size_t)(nbase + n) * ld + kbase;
#pragma unroll
  for (int r = 0; r < 8; ++r)
    f.u[r] = *(const uint32_t*)(p + (r >> 2) * 16 + h * 8 + (r & 3) * 2);
  return f.v;
}

__device__ __forceinline__ v8f wmma_f16(v16h a, v16h b, v8f c) {
  return __builtin_amdgcn_wmma_f32_16x16x32_f16(false, a, false, b, (short)0, c, false, false);
}

// Async copy of one 16-byte segment global -> LDS (ASYNCcnt-tracked DMA path).
__device__ __forceinline__ void async_copy_b128(uint32_t lds_off, const void* gptr) {
  asm volatile("global_load_async_to_lds_b128 %0, %1, off"
               :: "v"(lds_off), "v"(gptr) : "memory");
}
// Async loads complete in order: waiting ASYNCcnt<=2 retires the older tile's
// pair while the newer tile's pair is still in flight (double buffering).
__device__ __forceinline__ void wait_async0() {
  asm volatile("s_wait_asynccnt 0x0" ::: "memory");
}
__device__ __forceinline__ void wait_async2() {
  asm volatile("s_wait_asynccnt 0x2" ::: "memory");
}

// ---------------------------------------------------------------- kernel 1
// RoPE on hidden (per head, half-dim 32) -> q16 (f16); plain f16 copy -> x16.
__global__ void rope_pack_kernel(const float* __restrict__ hidden,
                                 _Float16* __restrict__ q16,
                                 _Float16* __restrict__ x16) {
  int t = blockIdx.x * blockDim.x + threadIdx.x;
  int i = t & 31; t >>= 5;
  int n = t % NHEAD; t /= NHEAD;
  int s = t % S_LEN;
  int b = t / S_LEN;
  if (b >= BATCH) return;
  size_t base = ((size_t)(b * S_LEN + s) * NHEAD + n) * HDIM;
  float x1 = hidden[base + i];
  float x2 = hidden[base + 32 + i];
  float inv = __expf(-(float)i * (logf(10000.0f) / 32.0f));
  float ang = (float)s * inv;
  float c = cosf(ang), sn = sinf(ang);
  q16[base + i]      = (_Float16)(x1 * c - x2 * sn);
  q16[base + 32 + i] = (_Float16)(x2 * c + x1 * sn);
  x16[base + i]      = (_Float16)x1;
  x16[base + 32 + i] = (_Float16)x2;
}

// ---------------------------------------------------------------- kernel 2
// 768x768 transpose + f32->f16: WT[c*768 + k] = W[k*768 + c]  (K-contiguous B storage)
__global__ void transpose768_kernel(const float* __restrict__ W, _Float16* __restrict__ WT) {
  int t = blockIdx.x * blockDim.x + threadIdx.x;
  int k = t % HID;
  int c = t / HID;
  if (c >= HID) return;
  WT[(size_t)c * HID + k] = (_Float16)W[(size_t)k * HID + c];
}

// ---------------------------------------------------------------- kernel 3
// QKV projection, 32x64 per wave (8 wmma / k-step), 4 waves per block covering
// 128 rows x one head.  Weight tile (64x32, 4KB) async-staged to LDS,
// double-buffered so the DMA for step i+1 overlaps the WMMAs of step i.
// transposed==0: Out[b][n][s][d] ; transposed==1: Out[b][n][d][s] (for V)
__global__ void __launch_bounds__(128)
qkv_gemm_kernel(const _Float16* __restrict__ Abase,
                const _Float16* __restrict__ WT,
                _Float16* __restrict__ Out, int transposed) {
  __shared__ _Float16 Wtile[2][64 * 32];       // [buf][col][k], K-contiguous
  int tid  = threadIdx.x;
  int lane = tid & 31;
  int wave = tid >> 5;
  int n  = blockIdx.x;                         // head
  int sb = blockIdx.y;                         // 128-row block
  int b  = blockIdx.z;
  int row_base = sb * 128 + wave * 32;
  const _Float16* A0 = Abase + ((size_t)b * S_LEN + row_base) * HID;
  const _Float16* A1 = A0 + 16 * HID;
  const _Float16* Wn = WT + (size_t)n * 64 * HID;
  uint32_t ldsb[2] = { (uint32_t)(size_t)(&Wtile[0][0]),
                       (uint32_t)(size_t)(&Wtile[1][0]) };

  // two 16B segments per thread per k-step: seg ids tid and tid+128 (256 total)
  int r0 = tid >> 2,         o0 = (tid & 3) * 16;
  int r1 = (tid + 128) >> 2, o1 = ((tid + 128) & 3) * 16;

  // prologue: stage first tile into buffer 0
  async_copy_b128(ldsb[0] + (uint32_t)(r0 * 64 + o0),
                  (const char*)(Wn + (size_t)r0 * HID) + o0);
  async_copy_b128(ldsb[0] + (uint32_t)(r1 * 64 + o1),
                  (const char*)(Wn + (size_t)r1 * HID) + o1);

  v8f c[2][4] = {};
  int buf = 0;
  for (int kk = 0; kk < HID; kk += 32, buf ^= 1) {
    if (kk + 32 < HID) {
      // issue next tile into the other buffer, then wait only for the old pair
      async_copy_b128(ldsb[buf ^ 1] + (uint32_t)(r0 * 64 + o0),
                      (const char*)(Wn + (size_t)r0 * HID + kk + 32) + o0);
      async_copy_b128(ldsb[buf ^ 1] + (uint32_t)(r1 * 64 + o1),
                      (const char*)(Wn + (size_t)r1 * HID + kk + 32) + o1);
      wait_async2();
    } else {
      wait_async0();
    }
    __syncthreads();                 // all waves' current tile landed
    const _Float16* Wt = &Wtile[buf][0];
    v16h a0 = load_frag_a(A0, HID, kk);
    v16h a1 = load_frag_a(A1, HID, kk);
#pragma unroll
    for (int cs = 0; cs < 4; ++cs) {
      v16h bm = load_frag_b_t(Wt, 32, 0, cs * 16);
      c[0][cs] = wmma_f16(a0, bm, c[0][cs]);
      c[1][cs] = wmma_f16(a1, bm, c[1][cs]);
    }
    __syncthreads();                 // reads done before this buf is reused
  }

  int row0 = row_base + ((lane >> 4) << 3);
#pragma unroll
  for (int rs = 0; rs < 2; ++rs) {
#pragma unroll
    for (int cs = 0; cs < 4; ++cs) {
      int d = cs * 16 + (lane & 15);
      if (!transposed) {
        _Float16* o = Out + ((size_t)b * NHEAD + n) * S_LEN * HDIM;
#pragma unroll
        for (int r = 0; r < 8; ++r)
          o[(size_t)(row0 + rs * 16 + r) * HDIM + d] = (_Float16)c[rs][cs][r];
      } else {
        _Float16* o = Out + (((size_t)b * NHEAD + n) * HDIM + d) * S_LEN;
#pragma unroll
        for (int r = 0; r < 8; ++r)
          o[row0 + rs * 16 + r] = (_Float16)c[rs][cs][r];
      }
    }
  }
}

// ---------------------------------------------------------------- kernel 4
// Sliding-window flash attention: one wave per (b, head, 16-query tile).
__global__ void __launch_bounds__(32)
attn_kernel(const _Float16* __restrict__ Q,
            const _Float16* __restrict__ K,
            const _Float16* __restrict__ Vt,
            _Float16* __restrict__ ctx) {
  __shared__ _Float16 Plds[16 * 32];     // P tile staging: C-layout -> A-layout
  int lane = threadIdx.x & 31;
  int qt = blockIdx.x, n = blockIdx.y, b = blockIdx.z;
  size_t bn = (size_t)b * NHEAD + n;
  const _Float16* Qb = Q  + bn * S_LEN * HDIM;
  const _Float16* Kb = K  + bn * S_LEN * HDIM;
  const _Float16* Vb = Vt + bn * HDIM * S_LEN;   // [d][s]
  int q0 = qt * 16;
  v16h aq0 = load_frag_a(Qb + (size_t)q0 * HDIM, HDIM, 0);
  v16h aq1 = load_frag_a(Qb + (size_t)q0 * HDIM, HDIM, 32);

  float m[8], l[8];
  v8f O[4] = {};
#pragma unroll
  for (int r = 0; r < 8; ++r) { m[r] = -3.0e30f; l[r] = 0.0f; }

  int klo = q0 - (WIN - 1); if (klo < 0) klo = 0;
  int khi = q0 + 15 + (WIN - 1); if (khi > S_LEN - 1) khi = S_LEN - 1;
  int row0 = (lane >> 4) << 3;
  int colk = lane & 15;
  const float scale = 0.125f;      // 1/sqrt(64)

  for (int kt = (klo & ~31); kt <= khi; kt += 32) {
    v8f s0 = {}, s1 = {};
    s0 = wmma_f16(aq0, load_frag_b_t(Kb + (size_t)kt * HDIM, HDIM, 0, 0),  s0);
    s0 = wmma_f16(aq1, load_frag_b_t(Kb + (size_t)kt * HDIM, HDIM, 32, 0), s0);
    s1 = wmma_f16(aq0, load_frag_b_t(Kb + (size_t)kt * HDIM, HDIM, 0, 16),  s1);
    s1 = wmma_f16(aq1, load_frag_b_t(Kb + (size_t)kt * HDIM, HDIM, 32, 16), s1);

    int k0 = kt + colk, k1 = kt + 16 + colk;
    float p0[8], p1[8], red[8];
#pragma unroll
    for (int r = 0; r < 8; ++r) {
      int qr = q0 + row0 + r;
      float v0 = s0[r] * scale;
      float v1 = s1[r] * scale;
      int d0 = qr - k0; if (d0 < 0) d0 = -d0;
      int d1 = qr - k1; if (d1 < 0) d1 = -d1;
      if (d0 >= WIN || k0 >= S_LEN) v0 = -1.0e9f;   // band + overrun mask
      if (d1 >= WIN || k1 >= S_LEN) v1 = -1.0e9f;
      p0[r] = v0; p1[r] = v1;
      red[r] = fmaxf(v0, v1);
    }
#pragma unroll
    for (int r = 0; r < 8; ++r)
#pragma unroll
      for (int off = 1; off < 16; off <<= 1)
        red[r] = fmaxf(red[r], __shfl_xor(red[r], off, 32));

    float sc[8];
#pragma unroll
    for (int r = 0; r < 8; ++r) {
      float mn = fmaxf(m[r], red[r]);
      sc[r] = __expf(m[r] - mn);
      m[r] = mn;
      p0[r] = __expf(p0[r] - mn);
      p1[r] = __expf(p1[r] - mn);
      red[r] = p0[r] + p1[r];
    }
#pragma unroll
    for (int r = 0; r < 8; ++r) {
#pragma unroll
      for (int off = 1; off < 16; off <<= 1)
        red[r] += __shfl_xor(red[r], off, 32);
      l[r] = l[r] * sc[r] + red[r];
    }
#pragma unroll
    for (int f = 0; f < 4; ++f)
#pragma unroll
      for (int r = 0; r < 8; ++r)
        O[f][r] *= sc[r];

    __syncthreads();
#pragma unroll
    for (int r = 0; r < 8; ++r) {
      Plds[(row0 + r) * 32 + colk]      = (_Float16)p0[r];
      Plds[(row0 + r) * 32 + 16 + colk] = (_Float16)p1[r];
    }
    __syncthreads();
    v16h ap = load_frag_a(Plds, 32, 0);

#pragma unroll
    for (int f = 0; f < 4; ++f)
      O[f] = wmma_f16(ap, load_frag_b_t(Vb + kt, S_LEN, 0, f * 16), O[f]);
  }

  float rl[8];
#pragma unroll
  for (int r = 0; r < 8; ++r) rl[r] = 1.0f / l[r];
  _Float16* cb = ctx + (size_t)b * S_LEN * HID;
#pragma unroll
  for (int f = 0; f < 4; ++f) {
    int d = f * 16 + colk;
#pragma unroll
    for (int r = 0; r < 8; ++r) {
      int s = q0 + row0 + r;
      cb[(size_t)s * HID + n * HDIM + d] = (_Float16)(O[f][r] * rl[r]);
    }
  }
}

// ---------------------------------------------------------------- kernel 5
// out(8192x768 f32) = ctx(8192x768 f16) @ Wo; 32x64 per wave,
// double-buffered LDS-staged WoT.
__global__ void __launch_bounds__(128)
out_gemm_kernel(const _Float16* __restrict__ Actx,
                const _Float16* __restrict__ WoT,
                float* __restrict__ out) {
  __shared__ _Float16 Wtile[2][64 * 32];
  int tid  = threadIdx.x;
  int lane = tid & 31;
  int wave = tid >> 5;
  int cb = blockIdx.x;                         // 64-col block (0..11)
  int sb = blockIdx.y;                         // 128-row block (0..63)
  int row_base = sb * 128 + wave * 32;
  const _Float16* A0 = Actx + (size_t)row_base * HID;
  const _Float16* A1 = A0 + 16 * HID;
  const _Float16* Wn = WoT + (size_t)cb * 64 * HID;
  uint32_t ldsb[2] = { (uint32_t)(size_t)(&Wtile[0][0]),
                       (uint32_t)(size_t)(&Wtile[1][0]) };

  int r0 = tid >> 2,         o0 = (tid & 3) * 16;
  int r1 = (tid + 128) >> 2, o1 = ((tid + 128) & 3) * 16;

  async_copy_b128(ldsb[0] + (uint32_t)(r0 * 64 + o0),
                  (const char*)(Wn + (size_t)r0 * HID) + o0);
  async_copy_b128(ldsb[0] + (uint32_t)(r1 * 64 + o1),
                  (const char*)(Wn + (size_t)r1 * HID) + o1);

  v8f c[2][4] = {};
  int buf = 0;
  for (int kk = 0; kk < HID; kk += 32, buf ^= 1) {
    if (kk + 32 < HID) {
      async_copy_b128(ldsb[buf ^ 1] + (uint32_t)(r0 * 64 + o0),
                      (const char*)(Wn + (size_t)r0 * HID + kk + 32) + o0);
      async_copy_b128(ldsb[buf ^ 1] + (uint32_t)(r1 * 64 + o1),
                      (const char*)(Wn + (size_t)r1 * HID + kk + 32) + o1);
      wait_async2();
    } else {
      wait_async0();
    }
    __syncthreads();
    const _Float16* Wt = &Wtile[buf][0];
    v16h a0 = load_frag_a(A0, HID, kk);
    v16h a1 = load_frag_a(A1, HID, kk);
#pragma unroll
    for (int cs = 0; cs < 4; ++cs) {
      v16h bm = load_frag_b_t(Wt, 32, 0, cs * 16);
      c[0][cs] = wmma_f16(a0, bm, c[0][cs]);
      c[1][cs] = wmma_f16(a1, bm, c[1][cs]);
    }
    __syncthreads();
  }

  int row0 = row_base + ((lane >> 4) << 3);
#pragma unroll
  for (int rs = 0; rs < 2; ++rs)
#pragma unroll
    for (int cs = 0; cs < 4; ++cs) {
      int col = cb * 64 + cs * 16 + (lane & 15);
#pragma unroll
      for (int r = 0; r < 8; ++r)
        out[(size_t)(row0 + rs * 16 + r) * HID + col] = c[rs][cs][r];
    }
}

// ----------------------------------------------------------------
extern "C" void kernel_launch(void* const* d_in, const int* in_sizes, int n_in,
                              void* d_out, int out_size, void* d_ws, size_t ws_size,
                              hipStream_t stream) {
  (void)in_sizes; (void)n_in; (void)out_size; (void)ws_size;
  const float* hidden = (const float*)d_in[0];
  const float* Wq = (const float*)d_in[1];
  const float* Wk = (const float*)d_in[2];
  const float* Wv = (const float*)d_in[3];
  const float* Wo = (const float*)d_in[4];
  float* out = (float*)d_out;

  _Float16* ws = (_Float16*)d_ws;
  const size_t NBS = (size_t)BATCH * S_LEN * HID;   // 6,291,456
  const size_t NW  = (size_t)HID * HID;             //   589,824
  _Float16* q16 = ws;             // roped hidden, f16
  _Float16* x16 = q16 + NBS;      // raw hidden, f16
  _Float16* WqT = x16 + NBS;
  _Float16* WkT = WqT + NW;
  _Float16* WvT = WkT + NW;
  _Float16* WoT = WvT + NW;
  _Float16* Qh  = WoT + NW;       // [b][n][s][d]
  _Float16* Kh  = Qh + NBS;       // [b][n][s][d]
  _Float16* Vt  = Kh + NBS;       // [b][n][d][s]
  _Float16* ctx = Vt + NBS;       // [b][s][n*d]

  rope_pack_kernel<<<dim3((BATCH * S_LEN * NHEAD * 32) / 256), 256, 0, stream>>>(hidden, q16, x16);
  transpose768_kernel<<<dim3(NW / 256), 256, 0, stream>>>(Wq, WqT);
  transpose768_kernel<<<dim3(NW / 256), 256, 0, stream>>>(Wk, WkT);
  transpose768_kernel<<<dim3(NW / 256), 256, 0, stream>>>(Wv, WvT);
  transpose768_kernel<<<dim3(NW / 256), 256, 0, stream>>>(Wo, WoT);

  qkv_gemm_kernel<<<dim3(NHEAD, 16, BATCH), 128, 0, stream>>>(q16, WqT, Qh, 0);
  qkv_gemm_kernel<<<dim3(NHEAD, 16, BATCH), 128, 0, stream>>>(q16, WkT, Kh, 0);
  qkv_gemm_kernel<<<dim3(NHEAD, 16, BATCH), 128, 0, stream>>>(x16, WvT, Vt, 1);

  attn_kernel<<<dim3(128, NHEAD, BATCH), 32, 0, stream>>>(Qh, Kh, Vt, ctx);

  out_gemm_kernel<<<dim3(12, 64), 128, 0, stream>>>(ctx, WoT, out);
}
